// MomentumContrastiveLoss_5566277616543
// MI455X (gfx1250) — matrix operand order repaired and checked by reading, not compiled, not run on hardware
//
#include <hip/hip_runtime.h>

#define B_N 1024
#define D_N 2048
#define Q_N 8192
#define TAU_INV 20.0f
#define S_SC 64.0f
#define EPS_L 1e-10f

// GEMM tiling
#define BM 64
#define BN 128
#define BK 64
#define AST 72   // padded LDS row stride (bf16 elems): 144B rows -> 16B aligned, banks spread
#define KST 72

typedef __attribute__((ext_vector_type(16))) __bf16 v16bf;
typedef __attribute__((ext_vector_type(8)))  float  v8f;

union FragBF { uint4 q[2]; v16bf v; };

__device__ __forceinline__ unsigned short f2bf(float x) {
    unsigned u = __float_as_uint(x);
    unsigned r = u + 0x7FFFu + ((u >> 16) & 1u);   // round-to-nearest-even
    return (unsigned short)(r >> 16);
}

// CDNA5 async DMA: global -> LDS, 16B per lane, tracked by ASYNCcnt.
// lds_off = wave-relative LDS byte address (low 32 bits of generic LDS pointer),
// g = per-lane 64-bit global address (GV mode, SADDR=off).
__device__ __forceinline__ void g2l_async_b128(unsigned lds_off, const void* g) {
    asm volatile("global_load_async_to_lds_b128 %0, %1, off"
                 :: "v"(lds_off), "v"(g)
                 : "memory");
}

__device__ __forceinline__ void wait_async_le(int n) {
#if __has_builtin(__builtin_amdgcn_s_wait_asynccnt)
    if (n == 0) __builtin_amdgcn_s_wait_asynccnt(0);
    else        __builtin_amdgcn_s_wait_asynccnt(6);
#else
    if (n == 0) asm volatile("s_wait_asynccnt 0x0" ::: "memory");
    else        asm volatile("s_wait_asynccnt 0x6" ::: "memory");
#endif
}

// ---------------------------------------------------------------------------
// 1) Row norms + bf16 conversion + merged queue labels
//    blocks [0,B): A rows from dq_g; blocks [B,B+Q): K rows from dp_g/queue
// ---------------------------------------------------------------------------
__global__ void __launch_bounds__(256)
mcl_prep_kernel(const float* __restrict__ dq, const float* __restrict__ dp,
                const float* __restrict__ qfeat,
                const int* __restrict__ labels, const int* __restrict__ qlabels,
                unsigned short* __restrict__ Abf, unsigned short* __restrict__ Kbf,
                float* __restrict__ inv_a, float* __restrict__ inv_k,
                int* __restrict__ ql)
{
    __shared__ float red[256];
    const int r = blockIdx.x;
    const int t = threadIdx.x;

    const float* src;
    unsigned short* dst;
    float* invp;
    if (r < B_N) {
        src  = dq + (size_t)r * D_N;
        dst  = Abf + (size_t)r * D_N;
        invp = inv_a + r;
    } else {
        const int j = r - B_N;
        src  = (j < B_N) ? (dp + (size_t)j * D_N)
                         : (qfeat + (size_t)(j - B_N) * D_N);
        dst  = Kbf + (size_t)j * D_N;
        invp = inv_k + j;
        if (t == 0) ql[j] = (j < B_N) ? labels[j] : qlabels[j - B_N];
    }

    float ss = 0.f;
    for (int c = t; c < D_N / 4; c += 256) {
        float4 x = ((const float4*)src)[c];
        ss += x.x * x.x + x.y * x.y + x.z * x.z + x.w * x.w;
        ushort4 o;
        o.x = f2bf(x.x); o.y = f2bf(x.y); o.z = f2bf(x.z); o.w = f2bf(x.w);
        ((ushort4*)dst)[c] = o;
    }
    red[t] = ss; __syncthreads();
    for (int s = 128; s > 0; s >>= 1) {
        if (t < s) red[t] += red[t + s];
        __syncthreads();
    }
    if (t == 0) *invp = 1.0f / fmaxf(sqrtf(red[0]), 1e-12f);
}

// ---------------------------------------------------------------------------
// 2) cos = A_bf16 @ K_bf16^T, f32 accumulation, inverse-norm post-scale.
//    Macro tile 64x128, 8 waves (2x4), each wave 2x2 tiles of 16x16.
//    Double-buffered LDS staging via global_load_async_to_lds_b128 (ASYNCcnt).
// ---------------------------------------------------------------------------
__global__ void __launch_bounds__(256)
mcl_gemm_kernel(const unsigned short* __restrict__ Abf,
                const unsigned short* __restrict__ Kbf,
                const float* __restrict__ inv_a,
                const float* __restrict__ inv_k,
                float* __restrict__ cosm)
{
    __shared__ alignas(16) unsigned short As[2][BM * AST];
    __shared__ alignas(16) unsigned short Ks[2][BN * KST];

    const int t     = threadIdx.x;
    const int lane  = t & 31;
    const int wv    = t >> 5;
    const int wm    = wv >> 2;     // 0..1
    const int wn    = wv & 3;      // 0..3
    const int l16   = lane & 15;
    const int lhalf = lane >> 4;   // 0/1

    const int mBase = blockIdx.y * BM;
    const int nBase = blockIdx.x * BN;

    // Per-thread staging slots: 2 x A (64x64 tile), 4 x K (128x64 tile).
    // Each async b128 moves 16B/lane; per wave 6 instructions = 3KB, block = 24KB/tile.
    unsigned aLds[2]; const unsigned short* aG[2];
    unsigned kLds[4]; const unsigned short* kG[4];
    #pragma unroll
    for (int i = 0; i < 2; ++i) {
        const int idx = t + i * 256;
        const int row = idx >> 3;
        const int c8  = (idx & 7) * 8;
        aLds[i] = (unsigned)(row * AST + c8);                      // element offset in buffer
        aG[i]   = Abf + (size_t)(mBase + row) * D_N + c8;          // + kk at issue time
    }
    #pragma unroll
    for (int i = 0; i < 4; ++i) {
        const int idx = t + i * 256;
        const int row = idx >> 3;
        const int c8  = (idx & 7) * 8;
        kLds[i] = (unsigned)(row * KST + c8);
        kG[i]   = Kbf + (size_t)(nBase + row) * D_N + c8;
    }

    auto issue_tile = [&](int buf, int kk) {
        #pragma unroll
        for (int i = 0; i < 2; ++i)
            g2l_async_b128((unsigned)(size_t)&As[buf][aLds[i]], aG[i] + kk);
        #pragma unroll
        for (int i = 0; i < 4; ++i)
            g2l_async_b128((unsigned)(size_t)&Ks[buf][kLds[i]], kG[i] + kk);
    };

    v8f acc[2][2] = {};

    issue_tile(0, 0);                       // prologue: tile 0 -> buffer 0
    int ib = 0;
    for (int kk = 0; kk < D_N; kk += BK, ib ^= 1) {
        // Issue next tile into the other buffer (wrap-around dummy on last iter
        // keeps the loop branch-free; dangling loads retire at s_endpgm).
        const int knext = (kk + BK < D_N) ? (kk + BK) : 0;
        issue_tile(ib ^ 1, knext);

        // <=12 outstanding; async loads complete in order, so waiting to <=6
        // drains exactly the current tile's 6 transfers (per wave).
        wait_async_le(6);
        __syncthreads();                    // all waves' current tile landed

        const unsigned short* Asb = As[ib];
        const unsigned short* Ksb = Ks[ib];
        #pragma unroll
        for (int k2 = 0; k2 < BK; k2 += 32) {
            FragBF aF[2], bF[2];
            // A fragment: lane holds row M=l16, K = kb+{0..7} and kb+16+{0..7}, kb=lhalf*8
            #pragma unroll
            for (int tm = 0; tm < 2; ++tm) {
                const int rA = wm * 32 + tm * 16 + l16;
                const int kb = lhalf * 8;
                aF[tm].q[0] = *(const uint4*)&Asb[rA * AST + k2 + kb];
                aF[tm].q[1] = *(const uint4*)&Asb[rA * AST + k2 + kb + 16];
            }
            // B fragment: lane holds col N=l16, K = kb2+{0..7} and kb2+16+{0..7}, kb2=lhalf*8
            #pragma unroll
            for (int tn = 0; tn < 2; ++tn) {
                const int rB  = wn * 32 + tn * 16 + l16;
                const int kb2 = lhalf * 8;
                bF[tn].q[0] = *(const uint4*)&Ksb[rB * KST + k2 + kb2];
                bF[tn].q[1] = *(const uint4*)&Ksb[rB * KST + k2 + kb2 + 16];
            }
            #pragma unroll
            for (int tm = 0; tm < 2; ++tm)
                #pragma unroll
                for (int tn = 0; tn < 2; ++tn)
                    acc[tm][tn] = __builtin_amdgcn_wmma_f32_16x16x32_bf16(
                        false, aF[tm].v, false, bF[tn].v,
                        (short)0, acc[tm][tn], false, false);
        }
        __syncthreads();   // everyone done reading buf ib before it is re-issued
    }

    // epilogue: rank-1 inverse-norm scaling (exact normalization of f32 accum)
    #pragma unroll
    for (int tm = 0; tm < 2; ++tm) {
        #pragma unroll
        for (int tn = 0; tn < 2; ++tn) {
            const int n  = nBase + wn * 32 + tn * 16 + l16;
            const float ik = inv_k[n];
            #pragma unroll
            for (int v = 0; v < 8; ++v) {
                const int m = mBase + wm * 32 + tm * 16 + lhalf * 8 + v;
                cosm[(size_t)m * Q_N + n] = acc[tm][tn][v] * inv_a[m] * ik;
            }
        }
    }
}

// ---------------------------------------------------------------------------
// 3) CurricularFace margin logit scattered into cos[i, labels[i]]
// ---------------------------------------------------------------------------
__global__ void __launch_bounds__(256)
mcl_cf_kernel(const float* __restrict__ dq, const float* __restrict__ cfw,
              const int* __restrict__ labels, const float* __restrict__ inv_a,
              float* __restrict__ cosm)
{
    __shared__ float rd[256];
    __shared__ float rw[256];
    const int i = blockIdx.x;
    const int t = threadIdx.x;
    const int lab = labels[i];
    const float* x = dq  + (size_t)i   * D_N;
    const float* w = cfw + (size_t)lab * D_N;

    float dot = 0.f, wss = 0.f;
    for (int c = t; c < D_N / 4; c += 256) {
        float4 xv = ((const float4*)x)[c];
        float4 wv = ((const float4*)w)[c];
        dot += xv.x * wv.x + xv.y * wv.y + xv.z * wv.z + xv.w * wv.w;
        wss += wv.x * wv.x + wv.y * wv.y + wv.z * wv.z + wv.w * wv.w;
    }
    rd[t] = dot; rw[t] = wss; __syncthreads();
    for (int s = 128; s > 0; s >>= 1) {
        if (t < s) { rd[t] += rd[t + s]; rw[t] += rw[t + s]; }
        __syncthreads();
    }
    if (t == 0) {
        float tl = rd[0] * inv_a[i] / fmaxf(sqrtf(rw[0]), 1e-12f);
        tl = fminf(fmaxf(tl, -1.0f), 1.0f);
        const float cos_m = 0.87758256189f;   // cos(0.5)
        const float sin_m = 0.47942553860f;   // sin(0.5)
        const float thr   = -0.87758256189f;  // cos(pi-0.5)
        const float mm    = 0.23971276930f;   // sin(pi-0.5)*0.5
        float sin_t = sqrtf(fmaxf(1.0f - tl * tl, 0.0f));
        float ctm = tl * cos_m - sin_t * sin_m;
        float ft  = (tl > thr) ? ctm : (tl - mm);
        cosm[(size_t)i * Q_N + lab] = S_SC * ft;
    }
}

// ---------------------------------------------------------------------------
// 4) Per-row contrastive reduction (e-values cached in 32 regs + pos bitmask)
// ---------------------------------------------------------------------------
__global__ void __launch_bounds__(256)
mcl_rowloss_kernel(const float* __restrict__ cosm, const int* __restrict__ ql,
                   const int* __restrict__ labels,
                   float* __restrict__ row_loss, int* __restrict__ row_npos)
{
    __shared__ float sred[256];
    __shared__ int   ired[256];
    const int i = blockIdx.x;
    const int t = threadIdx.x;
    const int lab = labels[i];
    const float* row = cosm + (size_t)i * Q_N;

    float ev[32];
    unsigned mask = 0u;
    float negp = 0.f;
    int np = 0;
    #pragma unroll
    for (int c = 0; c < 32; ++c) {
        const int j = t + c * 256;
        const float e = __expf(row[j] * TAU_INV);
        ev[c] = e;
        if (ql[j] == lab) { mask |= (1u << c); ++np; }
        else              { negp += e; }
    }
    sred[t] = negp; ired[t] = np; __syncthreads();
    for (int s = 128; s > 0; s >>= 1) {
        if (t < s) { sred[t] += sred[t + s]; ired[t] += ired[t + s]; }
        __syncthreads();
    }
    const float neg = sred[0];
    const int npTot = ired[0];
    __syncthreads();

    float lp = 0.f;
    #pragma unroll
    for (int c = 0; c < 32; ++c) {
        if (mask & (1u << c)) {
            const float e = ev[c];
            lp += __logf(e + neg + EPS_L) - __logf(e + EPS_L);
        }
    }
    sred[t] = lp; __syncthreads();
    for (int s = 128; s > 0; s >>= 1) {
        if (t < s) sred[t] += sred[t + s];
        __syncthreads();
    }
    if (t == 0) { row_loss[i] = sred[0]; row_npos[i] = npTot; }
}

// ---------------------------------------------------------------------------
// 5) Final scalar: sum(row_loss) / max(sum(row_npos), 1)
// ---------------------------------------------------------------------------
__global__ void __launch_bounds__(256)
mcl_finalize_kernel(const float* __restrict__ row_loss,
                    const int* __restrict__ row_npos, float* __restrict__ out)
{
    __shared__ float sred[256];
    __shared__ int   ired[256];
    const int t = threadIdx.x;
    float ls = 0.f; int np = 0;
    for (int i = t; i < B_N; i += 256) { ls += row_loss[i]; np += row_npos[i]; }
    sred[t] = ls; ired[t] = np; __syncthreads();
    for (int s = 128; s > 0; s >>= 1) {
        if (t < s) { sred[t] += sred[t + s]; ired[t] += ired[t + s]; }
        __syncthreads();
    }
    if (t == 0) {
        const int n = ired[0] > 1 ? ired[0] : 1;
        out[0] = sred[0] / (float)n;
    }
}

extern "C" void kernel_launch(void* const* d_in, const int* in_sizes, int n_in,
                              void* d_out, int out_size, void* d_ws, size_t ws_size,
                              hipStream_t stream)
{
    const float* dq      = (const float*)d_in[0];   // [B,D]
    const float* dp      = (const float*)d_in[1];   // [B,D]
    const int*   labels  = (const int*)  d_in[2];   // [B]
    const float* qfeat   = (const float*)d_in[3];   // [Q,D]
    const int*   qlabels = (const int*)  d_in[4];   // [Q]
    const float* cfw     = (const float*)d_in[5];   // [C,D]

    char* ws = (char*)d_ws;
    size_t off = 0;
    auto carve = [&](size_t n) -> void* {
        void* p = ws + off;
        off += (n + 255) & ~((size_t)255);
        return p;
    };
    unsigned short* Abf   = (unsigned short*)carve((size_t)B_N * D_N * 2); //  4 MB
    unsigned short* Kbf   = (unsigned short*)carve((size_t)Q_N * D_N * 2); // 32 MB
    float* cosm           = (float*)carve((size_t)B_N * Q_N * 4);          // 32 MB
    float* inv_a          = (float*)carve((size_t)B_N * 4);
    float* inv_k          = (float*)carve((size_t)Q_N * 4);
    int*   ql             = (int*)  carve((size_t)Q_N * 4);
    float* row_loss       = (float*)carve((size_t)B_N * 4);
    int*   row_npos       = (int*)  carve((size_t)B_N * 4);
    (void)ws_size; (void)in_sizes; (void)n_in; (void)out_size;

    mcl_prep_kernel<<<B_N + Q_N, 256, 0, stream>>>(dq, dp, qfeat, labels, qlabels,
                                                   Abf, Kbf, inv_a, inv_k, ql);
    mcl_gemm_kernel<<<dim3(Q_N / BN, B_N / BM), 256, 0, stream>>>(Abf, Kbf, inv_a, inv_k, cosm);
    mcl_cf_kernel<<<B_N, 256, 0, stream>>>(dq, cfw, labels, inv_a, cosm);
    mcl_rowloss_kernel<<<B_N, 256, 0, stream>>>(cosm, ql, labels, row_loss, row_npos);
    mcl_finalize_kernel<<<1, 256, 0, stream>>>(row_loss, row_npos, (float*)d_out);
}
